// HilbertDilatedAttentionSimple_58926951301483
// MI455X (gfx1250) — matrix operand order repaired and checked by reading, not compiled, not run on hardware
//
#include <hip/hip_runtime.h>

typedef __attribute__((ext_vector_type(16))) __bf16 v16bf;
typedef __attribute__((ext_vector_type(8)))  __bf16 v8bf;
typedef __attribute__((ext_vector_type(4)))  __bf16 v4bf;
typedef __attribute__((ext_vector_type(8)))  float  v8f;
typedef __attribute__((ext_vector_type(4)))  int    v4i;

#define NDIM      1024
#define NHEADS    16
#define SEGLEN    64
#define NBATCH    4
#define SEQLEN    4096
#define NROWS     (NBATCH * SEQLEN)   // 16384
#define NSEGS     (SEQLEN / SEGLEN)   // 64
#define QKSCALE   0.125f              // 64^-0.5

// ---------------------------------------------------------------------------
// CDNA5 async global->LDS copies (ASYNCcnt-tracked), with safe fallbacks
// ---------------------------------------------------------------------------
#if defined(__has_builtin)
#if __has_builtin(__builtin_amdgcn_global_load_async_to_lds_b128)
#define USE_ASYNC_COPY 1
#endif
#endif

typedef v4i __attribute__((address_space(1)))* gv4i_p;   // global (AS1)
typedef v4i __attribute__((address_space(3)))* sv4i_p;   // LDS (AS3)

__device__ __forceinline__ void async_copy_b128(const void* gsrc, void* ldst) {
#ifdef USE_ASYNC_COPY
    __builtin_amdgcn_global_load_async_to_lds_b128((gv4i_p)gsrc, (sv4i_p)ldst, 0, 0);
#else
    *(uint4*)ldst = *(const uint4*)gsrc;
#endif
}

__device__ __forceinline__ void wait_async_copies() {
#ifdef USE_ASYNC_COPY
#if __has_builtin(__builtin_amdgcn_s_wait_asynccnt)
    __builtin_amdgcn_s_wait_asynccnt(0);
#else
    asm volatile("s_wait_asynccnt 0x0" ::: "memory");
#endif
#endif
}

// ---------------------------------------------------------------------------
// helpers
// ---------------------------------------------------------------------------
__device__ __forceinline__ v8f vzero8() {
    v8f z;
#pragma unroll
    for (int i = 0; i < 8; ++i) z[i] = 0.0f;
    return z;
}

__device__ __forceinline__ v16bf vcat(v8bf lo, v8bf hi) {
    v16bf r;
#pragma unroll
    for (int i = 0; i < 8; ++i) { r[i] = lo[i]; r[i + 8] = hi[i]; }
    return r;
}

// 16x32 bf16 A fragment from LDS (row-major, row stride ldm elements).
// ISA layout: lane<16 -> M=lane, K={0..7,16..23}; lane>=16 -> M=lane-16, K={8..15,24..31}
__device__ __forceinline__ v16bf load_a_frag(const __bf16* base, int ldm, int m0, int k0) {
    int lane = threadIdx.x & 31;
    int m  = m0 + (lane & 15);
    int kb = (lane < 16) ? 0 : 8;
    const __bf16* p = base + m * ldm + k0 + kb;
    v8bf lo = *(const v8bf*)(p);
    v8bf hi = *(const v8bf*)(p + 16);
    return vcat(lo, hi);
}

// 32x16 bf16 B fragment from LDS holding B transposed: Bt[N][K], row stride ldn.
// ISA layout: lane<16 -> N=lane, K={0..7,16..23} ; lane>=16 -> N=lane-16, K={8..15,24..31}
__device__ __forceinline__ v16bf load_bt_frag(const __bf16* base, int ldn, int n0, int k0) {
    int lane = threadIdx.x & 31;
    int n  = n0 + (lane & 15);
    int kb = (lane < 16) ? 0 : 8;
    const __bf16* p = base + n * ldn + k0 + kb;
    v8bf lo = *(const v8bf*)(p);
    v8bf hi = *(const v8bf*)(p + 16);
    return vcat(lo, hi);
}

__device__ __forceinline__ v8f wmma_bf16(v16bf a, v16bf b, v8f c) {
    return __builtin_amdgcn_wmma_f32_16x16x32_bf16(
        /*neg_a=*/false, a, /*neg_b=*/false, b,
        /*c_mod=*/(short)0, c, /*reuse_a=*/false, /*reuse_b=*/false);
}

// ---------------------------------------------------------------------------
// one-shot f32 -> bf16 conversion (bandwidth bound; removes all conversion
// work from the GEMM inner loops)
// ---------------------------------------------------------------------------
__global__ __launch_bounds__(256)
void cvt_f32_bf16_kernel(const float* __restrict__ in, __bf16* __restrict__ out) {
    int i = (blockIdx.x * 256 + threadIdx.x) * 4;
    float4 v = *(const float4*)(in + i);
    v4bf o;
    o[0] = (__bf16)v.x; o[1] = (__bf16)v.y; o[2] = (__bf16)v.z; o[3] = (__bf16)v.w;
    *(v4bf*)(out + i) = o;
}

// ---------------------------------------------------------------------------
// GEMM: C[M,N] = A[M,K] * B[K,N]; A,B bf16, C f32 or bf16.
// block tile 128(M) x 128(N), BK = 32, 256 threads = 8 waves (4 M x 2 N),
// wave tile 32(M) x 64(N) = 8 WMMA per K step (2 A-frags x 4 B-frags).
// Double-buffered LDS: A tile staged with async global->LDS b128 copies,
// B tile staged manually (2-byte transpose) while WMMAs of the current
// buffer are in flight; s_wait_asynccnt + barrier closes each stage.
// ---------------------------------------------------------------------------
template <bool OUT_BF16>
__global__ __launch_bounds__(256)
void gemm_wmma_kernel(const __bf16* __restrict__ Ap, const __bf16* __restrict__ Bp,
                      void* __restrict__ Cp, int K, int N) {
    __shared__ __align__(16) __bf16 As[2][128 * 40];  // [M][K], row stride 40 (pad)
    __shared__ __align__(16) __bf16 Bt[2][128 * 40];  // transposed B tile [N][K]

    const int tid  = threadIdx.x;
    const int nblk = blockIdx.x * 128;
    const int mblk = blockIdx.y * 128;
    const int wid  = tid >> 5;
    const int wm   = wid & 3;       // 0..3 -> M offset wm*32
    const int wn   = wid >> 2;      // 0..1 -> N offset wn*64

    // ---- stagers -----------------------------------------------------------
    auto stageA = [&](int k0, int buf) {
        // 128 rows x 32 cols bf16 = 512 x 16B chunks; 2 async b128 per thread
#pragma unroll
        for (int it = 0; it < 2; ++it) {
            int t4 = tid + it * 256;
            int r  = t4 >> 2;            // 4 chunks per row
            int cv = (t4 & 3) << 3;      // col in elements
            async_copy_b128(Ap + (size_t)(mblk + r) * K + k0 + cv,
                            &As[buf][r * 40 + cv]);
        }
    };
    auto stageB = [&](int k0, int buf) {
        // 32(K) x 128(N) bf16, transposed into LDS: v8 loads + 2-byte scatter
#pragma unroll
        for (int it = 0; it < 2; ++it) {
            int t4 = tid + it * 256;
            int kk = t4 >> 4;            // 16 chunks per K-row of 128
            int nv = (t4 & 15) << 3;
            v8bf v = *(const v8bf*)(Bp + (size_t)(k0 + kk) * N + nblk + nv);
#pragma unroll
            for (int c = 0; c < 8; ++c) Bt[buf][(nv + c) * 40 + kk] = v[c];
        }
    };

    v8f acc[2][4];
#pragma unroll
    for (int i = 0; i < 2; ++i)
#pragma unroll
        for (int j = 0; j < 4; ++j) acc[i][j] = vzero8();

    // ---- prologue ----------------------------------------------------------
    stageA(0, 0);
    stageB(0, 0);
    wait_async_copies();
    __syncthreads();

    // ---- main pipeline -----------------------------------------------------
    for (int k0 = 0; k0 < K; k0 += 32) {
        const int  buf  = (k0 >> 5) & 1;
        const bool more = (k0 + 32) < K;

        if (more) stageA(k0 + 32, buf ^ 1);   // async issue, overlaps compute

        v16bf a0 = load_a_frag(As[buf], 40, wm * 32, 0);
        v16bf a1 = load_a_frag(As[buf], 40, wm * 32 + 16, 0);
        v16bf b0 = load_bt_frag(Bt[buf], 40, wn * 64, 0);
        v16bf b1 = load_bt_frag(Bt[buf], 40, wn * 64 + 16, 0);
        v16bf b2 = load_bt_frag(Bt[buf], 40, wn * 64 + 32, 0);
        v16bf b3 = load_bt_frag(Bt[buf], 40, wn * 64 + 48, 0);

        acc[0][0] = wmma_bf16(a0, b0, acc[0][0]);
        acc[0][1] = wmma_bf16(a0, b1, acc[0][1]);
        acc[0][2] = wmma_bf16(a0, b2, acc[0][2]);
        acc[0][3] = wmma_bf16(a0, b3, acc[0][3]);
        acc[1][0] = wmma_bf16(a1, b0, acc[1][0]);
        acc[1][1] = wmma_bf16(a1, b1, acc[1][1]);
        acc[1][2] = wmma_bf16(a1, b2, acc[1][2]);
        acc[1][3] = wmma_bf16(a1, b3, acc[1][3]);

        if (more) {
            stageB(k0 + 32, buf ^ 1);         // overlaps WMMA drain
            wait_async_copies();
        }
        __syncthreads();
    }

    // ---- epilogue: C-fragment layout -> global ----
    const int lane = tid & 31;
    const int ncol = lane & 15;
    const int mb   = (lane < 16) ? 0 : 8;
#pragma unroll
    for (int fm = 0; fm < 2; ++fm)
#pragma unroll
        for (int fn = 0; fn < 4; ++fn) {
            int gm0 = mblk + wm * 32 + fm * 16 + mb;
            int gn  = nblk + wn * 64 + fn * 16 + ncol;
#pragma unroll
            for (int i = 0; i < 8; ++i) {
                float vv = acc[fm][fn][i];
                if constexpr (OUT_BF16)
                    ((__bf16*)Cp)[(size_t)(gm0 + i) * N + gn] = (__bf16)vv;
                else
                    ((float*)Cp)[(size_t)(gm0 + i) * N + gn] = vv;
            }
        }
}

// ---------------------------------------------------------------------------
// Segmented Hilbert dilated attention. One block per (b, h, seg); 4 waves.
// qkv: bf16 [NROWS][3072] (q | k | v, head h at col t*1024 + h*64).
// Gather via hilbert map on load; scatter back via hilbert map on store
// (equivalent to the reference's argsort inverse permutation).
// ---------------------------------------------------------------------------
__global__ __launch_bounds__(128)
void hilbert_attn_kernel(const __bf16* __restrict__ qkv, const int* __restrict__ hmap,
                         __bf16* __restrict__ outp) {
    __shared__ __align__(16) __bf16 Qs[64 * 72];   // [q_row][d]
    __shared__ __align__(16) __bf16 Ks[32 * 72];   // [key][d]  (== B^T for Q.K^T)
    __shared__ __align__(16) __bf16 Vt[64 * 40];   // [d][key]  (== B^T for P.V)
    __shared__ __align__(16) __bf16 Ps[64 * 40];   // [q_row][key] softmax probs
    __shared__ int rows[64];

    const int blk = blockIdx.x;       // b*1024 + h*64 + s
    const int s   = blk & 63;
    const int h   = (blk >> 6) & 15;
    const int b   = blk >> 10;
    const int tid = threadIdx.x;

    if (tid < 64) rows[tid] = hmap[s * 64 + tid];
    __syncthreads();

    // Q gather: 64 rows x 64 d
#pragma unroll
    for (int it = 0; it < 4; ++it) {
        int t4 = tid + it * 128;
        int r  = t4 >> 3;
        int dv = (t4 & 7) << 3;
        size_t src = (size_t)(b * SEQLEN + rows[r]) * 3072 + h * 64 + dv;
        *(v8bf*)&Qs[r * 72 + dv] = *(const v8bf*)(qkv + src);
    }
    // K gather (dilated, stride 2): 32 rows x 64 d
#pragma unroll
    for (int it = 0; it < 2; ++it) {
        int t4 = tid + it * 128;
        int r  = t4 >> 3;
        int dv = (t4 & 7) << 3;
        size_t src = (size_t)(b * SEQLEN + rows[r * 2]) * 3072 + 1024 + h * 64 + dv;
        *(v8bf*)&Ks[r * 72 + dv] = *(const v8bf*)(qkv + src);
    }
    // V gather (dilated), stored transposed [d][key]
#pragma unroll
    for (int it = 0; it < 2; ++it) {
        int t4 = tid + it * 128;
        int j  = t4 >> 3;
        int dv = (t4 & 7) << 3;
        v8bf v = *(const v8bf*)(qkv + (size_t)(b * SEQLEN + rows[j * 2]) * 3072 + 2048 + h * 64 + dv);
#pragma unroll
        for (int c = 0; c < 8; ++c) Vt[(dv + c) * 40 + j] = v[c];
    }
    __syncthreads();

    const int w    = tid >> 5;
    const int m0   = w * 16;          // this wave's 16 query rows
    const int lane = tid & 31;

    // S = Q K'^T  (16 x 32 keys per wave)
    v8f s0 = vzero8(), s1 = vzero8();
#pragma unroll
    for (int k0 = 0; k0 < 64; k0 += 32) {
        v16bf aq = load_a_frag(Qs, 72, m0, k0);
        v16bf b0 = load_bt_frag(Ks, 72, 0, k0);
        v16bf b1 = load_bt_frag(Ks, 72, 16, k0);
        s0 = wmma_bf16(aq, b0, s0);
        s1 = wmma_bf16(aq, b1, s1);
    }

    // f32 softmax across 32 keys: per C-fragment VGPR i the 16 lanes of a half-
    // wave hold one query row's keys 0..15 (s0) and 16..31 (s1).
    float p0[8], p1[8];
#pragma unroll
    for (int i = 0; i < 8; ++i) {
        float x0 = s0[i] * QKSCALE, x1 = s1[i] * QKSCALE;
        float mx = fmaxf(x0, x1);
#pragma unroll
        for (int off = 8; off; off >>= 1) mx = fmaxf(mx, __shfl_xor(mx, off, 32));
        float e0 = __expf(x0 - mx), e1 = __expf(x1 - mx);
        float sm = e0 + e1;
#pragma unroll
        for (int off = 8; off; off >>= 1) sm += __shfl_xor(sm, off, 32);
        float inv = 1.0f / sm;
        p0[i] = e0 * inv;
        p1[i] = e1 * inv;
    }

    // re-layout P through LDS into A-fragment order
    {
        int key = lane & 15;
        int mb  = (lane < 16) ? 0 : 8;
#pragma unroll
        for (int i = 0; i < 8; ++i) {
            Ps[(m0 + mb + i) * 40 + key]      = (__bf16)p0[i];
            Ps[(m0 + mb + i) * 40 + key + 16] = (__bf16)p1[i];
        }
    }
    __syncthreads();

    // O = P V'  (16 x 64 per wave)
    v16bf ap = load_a_frag(Ps, 40, m0, 0);
    v8f o[4];
#pragma unroll
    for (int nt = 0; nt < 4; ++nt) {
        o[nt] = vzero8();
        v16bf bv = load_bt_frag(Vt, 40, nt * 16, 0);
        o[nt] = wmma_bf16(ap, bv, o[nt]);
    }

    // scatter back through hilbert map; layout [NROWS][1024] bf16
    {
        int ncol = lane & 15;
        int mb   = (lane < 16) ? 0 : 8;
#pragma unroll
        for (int nt = 0; nt < 4; ++nt)
#pragma unroll
            for (int i = 0; i < 8; ++i) {
                int gr = rows[m0 + mb + i];
                outp[(size_t)(b * SEQLEN + gr) * 1024 + h * 64 + nt * 16 + ncol] = (__bf16)o[nt][i];
            }
    }
}

// ---------------------------------------------------------------------------
// launch
// ---------------------------------------------------------------------------
extern "C" void kernel_launch(void* const* d_in, const int* in_sizes, int n_in,
                              void* d_out, int out_size, void* d_ws, size_t ws_size,
                              hipStream_t stream) {
    const float* x      = (const float*)d_in[0];
    const float* w_qkv  = (const float*)d_in[1];
    const float* w_proj = (const float*)d_in[2];
    const int*   hmap   = (const int*)d_in[3];

    // workspace layout (all bf16)
    __bf16* xb     = (__bf16*)d_ws;                       // [NROWS][1024]
    __bf16* wqkvb  = xb    + (size_t)NROWS * NDIM;        // [1024][3072]
    __bf16* wprojb = wqkvb + (size_t)NDIM * 3 * NDIM;     // [1024][1024]
    __bf16* qkv    = wprojb + (size_t)NDIM * NDIM;        // [NROWS][3072]
    __bf16* attn   = qkv    + (size_t)NROWS * 3 * NDIM;   // [NROWS][1024]
    float*  out    = (float*)d_out;                       // [NROWS][1024] f32

    // Phase 0: one-shot bf16 conversions (bandwidth bound)
    cvt_f32_bf16_kernel<<<((size_t)NROWS * NDIM) / 1024, 256, 0, stream>>>(x, xb);
    cvt_f32_bf16_kernel<<<((size_t)NDIM * 3 * NDIM) / 1024, 256, 0, stream>>>(w_qkv, wqkvb);
    cvt_f32_bf16_kernel<<<((size_t)NDIM * NDIM) / 1024, 256, 0, stream>>>(w_proj, wprojb);

    // Phase 1: QKV projection (bf16 x bf16 -> bf16)
    dim3 gA(3072 / 128, NROWS / 128);
    gemm_wmma_kernel<true><<<gA, 256, 0, stream>>>(xb, wqkvb, qkv, NDIM, 3 * NDIM);

    // Phase 2: segmented dilated attention with fused Hilbert (un)permute
    hilbert_attn_kernel<<<NBATCH * NHEADS * NSEGS, 128, 0, stream>>>(qkv, hmap, attn);

    // Phase 3: output projection (bf16 x bf16 -> f32)
    dim3 gC(1024 / 128, NROWS / 128);
    gemm_wmma_kernel<false><<<gC, 256, 0, stream>>>(attn, wprojb, out, NDIM, NDIM);
}
